// LigandGAT_28656021799364
// MI455X (gfx1250) — matrix operand order repaired and checked by reading, not compiled, not run on hardware
//
#include <hip/hip_runtime.h>

// ---------------------------------------------------------------------------
// LigandGAT for MI455X (gfx1250, wave32, WMMA + async-to-LDS).
// - K/V projections commute with the neighbor gather: compute Q|K|V once per
//   bond with bf16 WMMA (4x fewer FLOPs than reference dataflow); gather the
//   projected rows. bf16 K+V (205MB) stays mostly resident in the 192MB L2.
// - GEMM k-step batches all B-fragment loads + next-A prefetch, then uses a
//   sched_barrier so the scheduler can't re-sink loads into the WMMA burst
//   (round-2 asm showed s_wait_loadcnt 0x0 before every WMMA otherwise).
// - Attention K/V gather uses GLOBAL_LOAD_ASYNC_TO_LDS_B128 (ASYNCcnt path).
// ---------------------------------------------------------------------------

typedef __bf16 bf16_t;
typedef __attribute__((ext_vector_type(16))) __bf16 v16bf;
typedef __attribute__((ext_vector_type(8)))  __bf16 v8bf;
typedef __attribute__((ext_vector_type(8)))  float  v8f;

#define F_DIM     256
#define NHEADS    4
#define KEY_D     64
#define VAL_D     64
#define ITERS     6
#define N_ATOMS   100000
#define N_BONDS   200000
#define NEI       6
#define ATOM_FDIM 31
#define BOND_FDIM 37
#define NEG_SLOPE 0.1f
#define QKV_N     (3 * F_DIM)   // 768
#define KLAST     288           // 31 + 256 = 287, padded to 288 (16B-aligned rows)

__device__ __forceinline__ float leaky(float x) { return x > 0.f ? x : NEG_SLOPE * x; }

// ---------------------------------------------------------------------------
// Weight prep: transpose fp32 weights into bf16 [N][K] so WMMA B-fragments are
// contiguous 16B loads. Tiny (~0.6MB total), launched once per call.
// ---------------------------------------------------------------------------
__global__ void prep_qkvT(const float* __restrict__ Wq, const float* __restrict__ Wk,
                          const float* __restrict__ Wv, bf16_t* __restrict__ T) {
  int idx = blockIdx.x * 256 + threadIdx.x;       // 768*256
  int n = idx >> 8, k = idx & 255;
  float v = (n < 256) ? Wq[k * 256 + n]
          : (n < 512) ? Wk[k * 256 + (n - 256)]
                      : Wv[k * 256 + (n - 512)];
  T[n * 256 + k] = (bf16_t)v;
}

__global__ void prep_woutT(const float* __restrict__ Wout, float* __restrict__ Tf) {
  int idx = blockIdx.x * 256 + threadIdx.x;       // 256*64
  int n = idx >> 6, k = idx & 63;
  Tf[n * 64 + k] = Wout[k * 256 + n];             // Wout is [64,256]
}

__global__ void prep_lastT(const float* __restrict__ W_last, bf16_t* __restrict__ T) {
  int n = blockIdx.x;                             // 256 output features
  int kk = threadIdx.x;                           // 288 (padded K)
  T[n * KLAST + kk] = (kk < ATOM_FDIM + F_DIM) ? (bf16_t)W_last[kk * 256 + n] : (bf16_t)0.f;
}

// ---------------------------------------------------------------------------
// Bond embedding: fbonds[B,37] @ W_emb[37,256] + b_emb, leaky-relu.
// Keeps fp32 copy (residual term reused every iteration) + bf16 message.
// ---------------------------------------------------------------------------
__global__ void __launch_bounds__(256)
embed_kernel(const float* __restrict__ fbonds, const float* __restrict__ W_emb,
             const float* __restrict__ b_emb, float* __restrict__ fbonds_input,
             bf16_t* __restrict__ message) {
  __shared__ float sf[BOND_FDIM];
  int b = blockIdx.x, t = threadIdx.x;
  if (t < BOND_FDIM) sf[t] = fbonds[b * BOND_FDIM + t];
  __syncthreads();
  float acc = b_emb[t];
#pragma unroll
  for (int j = 0; j < BOND_FDIM; ++j) acc += sf[j] * W_emb[j * F_DIM + t];
  fbonds_input[(size_t)b * F_DIM + t] = acc;
  message[(size_t)b * F_DIM + t] = (bf16_t)leaky(acc);
}

// ---------------------------------------------------------------------------
// bf16 WMMA GEMM: C[M, NTPW*128] = A[M,K] * BT[N,K]^T.
// Block = 256 threads = 8 waves; each wave owns NTPW 16x16 f32 accumulators.
// Fragment layouts per CDNA5 ISA 7.12.2 (16-bit A 16x32, B 32x16, f32 C/D).
// Per k-step: batch all B-frag loads + next-A prefetch, sched_barrier, then
// WMMA burst with staggered s_wait_loadcnt instead of full drains.
// EPI 0: store raw bf16 (QKV). EPI 1: +bias, leaky-relu, store f32 (final).
// ---------------------------------------------------------------------------
__device__ __forceinline__ v16bf load_a_frag(const bf16_t* p) {   // chunks +0..7, +16..23
  const v8bf* q = (const v8bf*)p;
  v8bf lo = q[0], hi = q[2];
  v16bf r;
#pragma unroll
  for (int i = 0; i < 8; ++i) { r[i] = lo[i]; r[i + 8] = hi[i]; }
  return r;
}
__device__ __forceinline__ v16bf load_b_frag(const bf16_t* p) {   // chunks +0..7, +16..23
  const v8bf* q = (const v8bf*)p;
  v8bf lo = q[0], hi = q[2];
  v16bf r;
#pragma unroll
  for (int i = 0; i < 8; ++i) { r[i] = lo[i]; r[i + 8] = hi[i]; }
  return r;
}

template <int K, int NTPW, int EPI>
__global__ void __launch_bounds__(256)
wmma_gemm_kernel(const bf16_t* __restrict__ A, const bf16_t* __restrict__ BT,
                 bf16_t* __restrict__ Cb, float* __restrict__ Cf,
                 const float* __restrict__ bias) {
  const int N = NTPW * 128;
  int lane = threadIdx.x & 31;
  int wave = threadIdx.x >> 5;
  size_t row0 = (size_t)blockIdx.x * 16;

  // A-fragment addressing (lane<16: row=lane, K {0..7,16..23}; lane>=16: row=lane-16, K {8..15,24..31})
  const bf16_t* Arow = A + (row0 + (lane & 15)) * K + (lane >> 4) * 8;
  // B-fragment addressing (lane holds one column, K {0..7,16..23}; lanes>=16 take K {8..15,24..31})
  int cidx = lane & 15;
  int bkoff = (lane >> 4) * 8;

  v8f acc[NTPW];
#pragma unroll
  for (int j = 0; j < NTPW; ++j)
#pragma unroll
    for (int i = 0; i < 8; ++i) acc[j][i] = 0.f;

  v16bf a_cur = load_a_frag(Arow);
#pragma unroll
  for (int kt = 0; kt < K; kt += 32) {
    // Batch all B fragments for this k-step (one clause group, staggered waits).
    v16bf b[NTPW];
#pragma unroll
    for (int j = 0; j < NTPW; ++j)
      b[j] = load_b_frag(BT + (size_t)((wave + 8 * j) * 16 + cidx) * K + kt + bkoff);
    // Prefetch next A fragment during the WMMA burst.
    v16bf a_nxt = (kt + 32 < K) ? load_a_frag(Arow + kt + 32) : a_cur;
    // Fence the scheduler: loads stay above, WMMAs below (no re-sinking).
    __builtin_amdgcn_sched_barrier(0);
#pragma unroll
    for (int j = 0; j < NTPW; ++j)
      acc[j] = __builtin_amdgcn_wmma_f32_16x16x32_bf16(
          /*neg_a=*/false, a_cur, /*neg_b=*/false, b[j],
          /*c_mod=*/(short)0, acc[j], /*reuse_a=*/false, /*reuse_b=*/false);
    a_cur = a_nxt;
  }

  // C/D layout: VGPR r -> M=r (+8 for lanes 16-31), N = lane&15
  int crow = (lane >> 4) * 8;
#pragma unroll
  for (int j = 0; j < NTPW; ++j) {
    int col = (wave + 8 * j) * 16 + cidx;
#pragma unroll
    for (int r = 0; r < 8; ++r) {
      size_t idx = (row0 + r + crow) * (size_t)N + col;
      if (EPI == 0) {
        Cb[idx] = (bf16_t)acc[j][r];
      } else {
        Cf[idx] = leaky(acc[j][r] + bias[col]);
      }
    }
  }
}

// ---------------------------------------------------------------------------
// Per-bond attention over 6 neighbors + Wout projection + residual + leaky.
// K/V rows (512B contiguous bf16, mostly L2 hits) are gathered with
// GLOBAL_LOAD_ASYNC_TO_LDS_B128: wave n moves neighbor n's K and V rows
// (32 lanes x 16B each), waits ASYNCcnt==0, then block-barriers.
// ---------------------------------------------------------------------------
__global__ void __launch_bounds__(256)
attention_kernel(const bf16_t* __restrict__ QKV, const int* __restrict__ bgraph,
                 const float* __restrict__ fbonds_input, const float* __restrict__ WoutTf,
                 bf16_t* __restrict__ message) {
  __shared__ float s_q[F_DIM];
  __shared__ __attribute__((aligned(16))) bf16_t s_kraw[NEI][F_DIM];
  __shared__ __attribute__((aligned(16))) bf16_t s_vraw[NEI][F_DIM];
  __shared__ float s_attn[NHEADS][NEI];
  __shared__ float s_out[VAL_D];
  __shared__ int   s_idx[NEI];

  int b = blockIdx.x, t = threadIdx.x;
  int wave = t >> 5, lane = t & 31;
  if (t < NEI) s_idx[t] = bgraph[b * NEI + t];
  __syncthreads();

  s_q[t] = (float)QKV[(size_t)b * QKV_N + t];

  if (wave < NEI) {   // wave n gathers neighbor n's K row + V row straight into LDS
    size_t base = (size_t)s_idx[wave] * QKV_N;
    unsigned long long gk = (unsigned long long)(uintptr_t)(QKV + base + F_DIM) + (unsigned)lane * 16u;
    unsigned long long gv = (unsigned long long)(uintptr_t)(QKV + base + 2 * F_DIM) + (unsigned)lane * 16u;
    unsigned int lk = (unsigned int)(uintptr_t)(&s_kraw[wave][0]) + (unsigned)lane * 16u;
    unsigned int lv = (unsigned int)(uintptr_t)(&s_vraw[wave][0]) + (unsigned)lane * 16u;
    asm volatile("global_load_async_to_lds_b128 %0, %1, off" :: "v"(lk), "v"(gk) : "memory");
    asm volatile("global_load_async_to_lds_b128 %0, %1, off" :: "v"(lv), "v"(gv) : "memory");
    asm volatile("s_wait_asynccnt 0x0" ::: "memory");  // LDS writes visible before barrier
  }
  __syncthreads();

  if (t < NHEADS) {                       // one thread per head: scores + softmax (NEI=6)
    float sc[NEI], mx = -3.0e38f;
#pragma unroll
    for (int n = 0; n < NEI; ++n) {
      float d = 0.f;
#pragma unroll
      for (int kk = 0; kk < KEY_D; ++kk)
        d += s_q[t * KEY_D + kk] * (float)s_kraw[n][t * KEY_D + kk];
      d *= 0.125f;                        // 1/sqrt(64)
      if (s_idx[n] == 0) d += -1e7f;      // padding-bond mask
      sc[n] = d;
      mx = fmaxf(mx, d);
    }
    float sum = 0.f;
#pragma unroll
    for (int n = 0; n < NEI; ++n) { sc[n] = __expf(sc[n] - mx); sum += sc[n]; }
    float inv = 1.f / sum;
#pragma unroll
    for (int n = 0; n < NEI; ++n) s_attn[t][n] = sc[n] * inv;
  }
  __syncthreads();

  if (t < VAL_D) {                        // attn-weighted V, mean over heads
    float o = 0.f;
#pragma unroll
    for (int h = 0; h < NHEADS; ++h)
#pragma unroll
      for (int n = 0; n < NEI; ++n) o += s_attn[h][n] * (float)s_vraw[n][h * VAL_D + t];
    s_out[t] = o * (1.f / NHEADS);
  }
  __syncthreads();

  // message = leaky(fbonds_input + out @ Wout)   (WoutTf is [256][64] row-major)
  float acc = fbonds_input[(size_t)b * F_DIM + t];
#pragma unroll
  for (int kk = 0; kk < VAL_D; ++kk) acc += s_out[kk] * WoutTf[t * VAL_D + kk];
  message[(size_t)b * F_DIM + t] = (bf16_t)leaky(acc);
}

// ---------------------------------------------------------------------------
// Per-atom gather-sum of incoming bond messages + concat with fatoms, packed
// bf16 into Xlast[A, 288] for the final WMMA GEMM.
// ---------------------------------------------------------------------------
__global__ void __launch_bounds__(256)
gather_concat_kernel(const bf16_t* __restrict__ message, const int* __restrict__ agraph,
                     const float* __restrict__ fatoms, bf16_t* __restrict__ Xlast) {
  __shared__ int sa[NEI];
  int a = blockIdx.x, t = threadIdx.x;
  if (t < NEI) sa[t] = agraph[a * NEI + t];
  __syncthreads();
  float ns = 0.f;
#pragma unroll
  for (int n = 0; n < NEI; ++n) ns += (float)message[(size_t)sa[n] * F_DIM + t];
  bf16_t* row = Xlast + (size_t)a * KLAST;
  row[ATOM_FDIM + t] = (bf16_t)ns;                 // cols 31..286
  if (t < ATOM_FDIM) row[t] = (bf16_t)fatoms[a * ATOM_FDIM + t];
  if (t == 0) row[KLAST - 1] = (bf16_t)0.f;        // pad col 287
}

// ---------------------------------------------------------------------------
extern "C" void kernel_launch(void* const* d_in, const int* in_sizes, int n_in,
                              void* d_out, int out_size, void* d_ws, size_t ws_size,
                              hipStream_t stream) {
  const float* fatoms  = (const float*)d_in[0];
  const float* fbonds  = (const float*)d_in[1];
  const int*   agraph  = (const int*)d_in[2];
  const int*   bgraph  = (const int*)d_in[3];
  // d_in[4] = lig_scope (unused by the reference computation)
  const float* W_emb   = (const float*)d_in[5];
  const float* b_emb   = (const float*)d_in[6];
  const float* Wq      = (const float*)d_in[7];
  const float* Wk      = (const float*)d_in[8];
  const float* Wv      = (const float*)d_in[9];
  const float* Wout    = (const float*)d_in[10];
  const float* W_last  = (const float*)d_in[11];
  const float* b_last  = (const float*)d_in[12];
  float* out = (float*)d_out;

  // Workspace carve (256B aligned slices)
  char* w = (char*)d_ws;
  auto carve = [&](size_t bytes) -> char* {
    char* p = w;
    w += (bytes + 255) & ~(size_t)255;
    return p;
  };
  float*  fbonds_input = (float*)carve((size_t)N_BONDS * F_DIM * 4);   // 204.8 MB
  bf16_t* message      = (bf16_t*)carve((size_t)N_BONDS * F_DIM * 2);  // 102.4 MB
  bf16_t* QKV          = (bf16_t*)carve((size_t)N_BONDS * QKV_N * 2);  // 307.2 MB
  bf16_t* WqkvT        = (bf16_t*)carve((size_t)QKV_N * F_DIM * 2);
  float*  WoutTf       = (float*)carve((size_t)F_DIM * VAL_D * 4);
  bf16_t* WlastT       = (bf16_t*)carve((size_t)F_DIM * KLAST * 2);
  bf16_t* Xlast        = (bf16_t*)carve((size_t)N_ATOMS * KLAST * 2);  // 57.6 MB

  prep_qkvT<<<QKV_N, 256, 0, stream>>>(Wq, Wk, Wv, WqkvT);
  prep_woutT<<<VAL_D, 256, 0, stream>>>(Wout, WoutTf);
  prep_lastT<<<F_DIM, KLAST, 0, stream>>>(W_last, WlastT);

  embed_kernel<<<N_BONDS, F_DIM, 0, stream>>>(fbonds, W_emb, b_emb, fbonds_input, message);

  for (int it = 0; it < ITERS - 1; ++it) {
    // Q|K|V = message @ Wqkv   (gather commutes with projection)
    wmma_gemm_kernel<F_DIM, 6, 0><<<N_BONDS / 16, 256, 0, stream>>>(
        message, WqkvT, QKV, nullptr, nullptr);
    attention_kernel<<<N_BONDS, F_DIM, 0, stream>>>(
        QKV, bgraph, fbonds_input, WoutTf, message);
  }

  gather_concat_kernel<<<N_ATOMS, F_DIM, 0, stream>>>(message, agraph, fatoms, Xlast);
  wmma_gemm_kernel<KLAST, 2, 1><<<N_ATOMS / 16, 256, 0, stream>>>(
      Xlast, WlastT, nullptr, out, b_last);
}
